// Hyper_GAT_Model_5360119186023
// MI455X (gfx1250) — matrix operand order repaired and checked by reading, not compiled, not run on hardware
//
#include <hip/hip_runtime.h>
#include <hip/hip_bf16.h>
#include <cstddef>

typedef __attribute__((ext_vector_type(16))) _Float16 v16h;
typedef __attribute__((ext_vector_type(8)))  _Float16 v8h;
typedef __attribute__((ext_vector_type(8)))  float    v8f;

#define BATCH 16
#define NWIN  30
#define NE    19
#define NF    128
#define NN    (NWIN * NE)       // 570 nodes per batch
#define HEADS 8
#define HID   80
#define DOUT  (HEADS * HID)     // 640
#define BN    (BATCH * NN)      // 9120 rows

// ---------------- workspace layout (bytes) ----------------
#define WS_NODES 0
#define WS_WL    2334720
#define WS_WR    2498560
#define WS_XL    2662400
#define WS_XR    26009600
#define WS_OUT   49356800

// ---------------- wave reductions (wave32) ----------------
__device__ __forceinline__ float wave_max(float v) {
    for (int o = 16; o > 0; o >>= 1) v = fmaxf(v, __shfl_xor(v, o, 32));
    return v;
}
__device__ __forceinline__ float wave_sum(float v) {
    for (int o = 16; o > 0; o >>= 1) v += __shfl_xor(v, o, 32);
    return v;
}

// ---------------- kernel 1: layout prep + fp16 conversion ----------------
__global__ __launch_bounds__(256)
void prep_kernel(const float* __restrict__ x,
                 const float* __restrict__ Wl, const float* __restrict__ Wr,
                 _Float16* __restrict__ nodes,
                 _Float16* __restrict__ wl16, _Float16* __restrict__ wr16) {
    int t = blockIdx.x * 256 + threadIdx.x;
    const int NODE_TOT = BN * NF;            // 1,167,360
    if (t < NODE_TOT) {
        int f  = t & (NF - 1);
        int r  = t >> 7;                     // b*570 + w*19 + e
        int e  = r % NE;
        int bw = r / NE;                     // b*30 + w
        nodes[t] = (_Float16)x[(size_t)(bw * NF + f) * NE + e];
    }
    int t2 = t - NODE_TOT;
    if (t2 >= 0 && t2 < DOUT * NF) {
        wl16[t2] = (_Float16)Wl[t2];
        wr16[t2] = (_Float16)Wr[t2];
    }
}

// ---------------- kernel 2: WMMA GEMM  XL/XR = nodes @ {Wl,Wr}^T + bias ----
// M = 9120, N = 640, K = 128.  One wave computes a 16x64 strip for BOTH
// weight matrices: the A fragment is loaded once per k-step and reused by
// 8 WMMAs (4 N-subtiles x {L,R}).  32 v_wmma per wave, load:wmma = 2.25.
__global__ __launch_bounds__(32)
void gemm_wmma_kernel(const _Float16* __restrict__ A,
                      const _Float16* __restrict__ WL, const _Float16* __restrict__ WR,
                      const float* __restrict__ bl, const float* __restrict__ br,
                      float* __restrict__ XL, float* __restrict__ XR) {
    const int tm   = blockIdx.x;             // 0..569  (16-row strip)
    const int cn0  = blockIdx.y * 64;        // 0..576  (64-col strip)
    const int lane = threadIdx.x;            // 0..31 (one wave)
    const int l15  = lane & 15;
    const int half = lane >> 4;              // 0 or 1

    // A fragment (16x32 f16): lane half 0 -> K [0..15], half 1 -> K [16..31]
    const size_t rowA = (size_t)(tm * 16 + l15) * NF;
    const int    aoff = half * 16;
    // B fragment (32x16 f16): lane half 0 -> K [0..15], half 1 -> K [16..31];
    // column n of B == row n of W (B = W^T), rows of W are K-contiguous.
    const int    boff = half * 16;

    union V16 { v16h v; v8h h[2]; };

    v8f accL[4] = {};
    v8f accR[4] = {};

#pragma unroll
    for (int kb = 0; kb < NF; kb += 32) {
        V16 ua;
        ua.h[0] = *(const v8h*)(A + rowA + kb + aoff);
        ua.h[1] = *(const v8h*)(A + rowA + kb + aoff + 8);
#pragma unroll
        for (int j = 0; j < 4; ++j) {
            const size_t rB = (size_t)(cn0 + j * 16 + l15) * NF;
            V16 ubL, ubR;
            ubL.h[0] = *(const v8h*)(WL + rB + kb + boff);
            ubL.h[1] = *(const v8h*)(WL + rB + kb + boff + 8);
            ubR.h[0] = *(const v8h*)(WR + rB + kb + boff);
            ubR.h[1] = *(const v8h*)(WR + rB + kb + boff + 8);
            accL[j] = __builtin_amdgcn_wmma_f32_16x16x32_f16(
                          false, ua.v, false, ubL.v, (short)0, accL[j], false, false);
            accR[j] = __builtin_amdgcn_wmma_f32_16x16x32_f16(
                          false, ua.v, false, ubR.v, (short)0, accR[j], false, false);
        }
    }

    // D layout: lane holds column n = lane&15; rows m = (lane>>4)*8 + r.
    const int m0 = tm * 16 + half * 8;
#pragma unroll
    for (int j = 0; j < 4; ++j) {
        const int   n   = cn0 + j * 16 + l15;
        const float bvL = bl[n];
        const float bvR = br[n];
#pragma unroll
        for (int r = 0; r < 8; ++r) {
            XL[(size_t)(m0 + r) * DOUT + n] = accL[j][r] + bvL;
            XR[(size_t)(m0 + r) * DOUT + n] = accR[j][r] + bvR;
        }
    }
}

// ---------------- kernel 3: fused GAT attention + aggregation + ELU --------
// One block per (b, dst-node) = 9120 blocks; 8 waves = 8 heads.
// In-neighbors of d = w*19+e: the 19 nodes of window w, plus same-electrode
// nodes in windows w-1 / w+1 when they exist (matches _build_edge_index).
__global__ __launch_bounds__(256)
void attn_kernel(const float* __restrict__ xl, const float* __restrict__ xr,
                 const float* __restrict__ att, const float* __restrict__ conv_bias,
                 float* __restrict__ out) {
    const int bd   = blockIdx.x;             // b*570 + d
    const int b    = bd / NN;
    const int d    = bd - b * NN;
    const int h    = threadIdx.x >> 5;       // head = wave id
    const int lane = threadIdx.x & 31;
    const int w    = d / NE;
    const int ns   = NE + (w > 0) + (w < NWIN - 1);

    const float* __restrict__ xr_d  = xr + ((size_t)bd * DOUT + h * HID);
    const float* __restrict__ att_h = att + h * HID;

    auto nbr = [&](int i) -> int {
        if (i < NE) return w * NE + i;
        if (w == 0) return d + NE;                  // only "next" exists
        return (i == NE) ? d - NE : d + NE;
    };

    // one lane per neighbor: e_i = sum_c lrelu(xl[s] + xr[d], 0.2) * att
    float e = -3.0e38f;
    if (lane < ns) {
        const int s = nbr(lane);
        const float* __restrict__ xl_s = xl + ((size_t)(b * NN + s) * DOUT + h * HID);
        float acc = 0.f;
        for (int c = 0; c < HID; ++c) {
            float v = xl_s[c] + xr_d[c];
            v = (v > 0.f) ? v : 0.2f * v;
            acc = fmaf(v, att_h[c], acc);
        }
        e = acc;
    }

    // wave-wide segment softmax (all 32 lanes execute the shuffles)
    const float m     = wave_max(e);
    const float ex    = (lane < ns) ? __expf(e - m) : 0.f;
    const float ssum  = wave_sum(ex);
    const float alpha = ex / (ssum + 1e-16f);

    // aggregation: agg[c] = sum_i alpha_i * xl[s_i, c]; lanes parallel over c.
    for (int c0 = 0; c0 < HID; c0 += 32) {
        const int  c   = c0 + lane;
        const bool act = (c < HID);
        float acc = 0.f;
        for (int i = 0; i < ns; ++i) {
            float a_i = __shfl(alpha, i, 32);       // uniform control flow here
            const int s = nbr(i);
            float xv = act ? xl[(size_t)(b * NN + s) * DOUT + h * HID + c] : 0.f;
            acc = fmaf(a_i, xv, acc);
        }
        if (act) {
            float o = acc + conv_bias[h * HID + c];
            o = (o > 0.f) ? o : (__expf(o) - 1.f);  // ELU
            out[(size_t)bd * DOUT + h * HID + c] = o;
        }
    }
}

// ---------------- kernel 4: attention pooling + FC ----------------
__global__ __launch_bounds__(256)
void pool_kernel(const float* __restrict__ out,
                 const float* __restrict__ Wpool, const float* __restrict__ bpool,
                 const float* __restrict__ Wfc,   const float* __restrict__ bfc,
                 float* __restrict__ logits) {
    __shared__ float sc[NN];
    __shared__ float red[256];
    const int b    = blockIdx.x;
    const int tid  = threadIdx.x;
    const int wv   = tid >> 5;
    const int lane = tid & 31;
    const float* __restrict__ ob = out + (size_t)b * NN * DOUT;

    // scores[n] = out[b,n,:] . Wpool + bpool  (wave per node)
    for (int n = wv; n < NN; n += 8) {
        const float* __restrict__ on = ob + (size_t)n * DOUT;
        float acc = 0.f;
        for (int c = lane; c < DOUT; c += 32) acc = fmaf(on[c], Wpool[c], acc);
        acc = wave_sum(acc);
        if (lane == 0) sc[n] = acc + bpool[0];
    }
    __syncthreads();

    // block max over N
    float m = -3.0e38f;
    for (int n = tid; n < NN; n += 256) m = fmaxf(m, sc[n]);
    red[tid] = m; __syncthreads();
    for (int s = 128; s > 0; s >>= 1) {
        if (tid < s) red[tid] = fmaxf(red[tid], red[tid + s]);
        __syncthreads();
    }
    m = red[0];
    __syncthreads();

    // exp + block sum
    float p = 0.f;
    for (int n = tid; n < NN; n += 256) {
        float e = __expf(sc[n] - m);
        sc[n] = e;
        p += e;
    }
    red[tid] = p; __syncthreads();
    for (int s = 128; s > 0; s >>= 1) {
        if (tid < s) red[tid] += red[tid + s];
        __syncthreads();
    }
    const float denom = red[0];
    __syncthreads();

    // pooled[c] = sum_n exp_n * out[n,c]; logit = (pooled . Wfc)/denom + bfc
    float part = 0.f;
    for (int c = tid; c < DOUT; c += 256) {
        float pc = 0.f;
        for (int n = 0; n < NN; ++n)
            pc = fmaf(sc[n], ob[(size_t)n * DOUT + c], pc);
        part = fmaf(pc, Wfc[c], part);
    }
    red[tid] = part; __syncthreads();
    for (int s = 128; s > 0; s >>= 1) {
        if (tid < s) red[tid] += red[tid + s];
        __syncthreads();
    }
    if (tid == 0) logits[b] = red[0] / denom + bfc[0];
}

// ---------------- host launcher ----------------
extern "C" void kernel_launch(void* const* d_in, const int* in_sizes, int n_in,
                              void* d_out, int out_size, void* d_ws, size_t ws_size,
                              hipStream_t stream) {
    (void)in_sizes; (void)n_in; (void)out_size; (void)ws_size;
    const float* x         = (const float*)d_in[0];
    /* d_in[1] = edge_index (int32) — structure is known & regular, unused */
    const float* Wl        = (const float*)d_in[2];
    const float* bl        = (const float*)d_in[3];
    const float* Wr        = (const float*)d_in[4];
    const float* br        = (const float*)d_in[5];
    const float* att       = (const float*)d_in[6];
    const float* conv_bias = (const float*)d_in[7];
    const float* Wpool     = (const float*)d_in[8];
    const float* bpool     = (const float*)d_in[9];
    const float* Wfc       = (const float*)d_in[10];
    const float* bfc       = (const float*)d_in[11];

    char* ws = (char*)d_ws;
    _Float16* nodes = (_Float16*)(ws + WS_NODES);
    _Float16* wl16  = (_Float16*)(ws + WS_WL);
    _Float16* wr16  = (_Float16*)(ws + WS_WR);
    float*    xl    = (float*)(ws + WS_XL);
    float*    xr    = (float*)(ws + WS_XR);
    float*    obuf  = (float*)(ws + WS_OUT);

    // 1) transpose + fp16 convert
    prep_kernel<<<4880, 256, 0, stream>>>(x, Wl, Wr, nodes, wl16, wr16);

    // 2) WMMA GEMMs: each wave does a 16x64 strip for BOTH Wl and Wr
    gemm_wmma_kernel<<<dim3(570, 10), 32, 0, stream>>>(nodes, wl16, wr16,
                                                       bl, br, xl, xr);

    // 3) fused attention (one block per (b, node), one wave per head)
    attn_kernel<<<BN, 256, 0, stream>>>(xl, xr, att, conv_bias, obuf);

    // 4) pooling + FC -> 16 logits
    pool_kernel<<<BATCH, 256, 0, stream>>>(obuf, Wpool, bpool, Wfc, bfc,
                                           (float*)d_out);
}
